// TCNGNNModel_34213709480009
// MI455X (gfx1250) — compile-verified, matrix operand order checked by reading, not run on hardware
//
#include <hip/hip_runtime.h>

// ---------------------------------------------------------------------------
// CDNA5 (gfx1250) fused TCN + GAT + MLP forward.
// All dense GEMMs use v_wmma_f32_16x16x32_bf16 (wave32, 16x16 tiles).
// Operands are pre-packed into the ISA 7.12.2 fragment layout:
//   lane = r + 16*group, element i covers kin = 8*group + (i&7) + 16*(i>>3)
// so each lane's 16-bf16 fragment is one contiguous 32B chunk (2x b128 loads).
// ---------------------------------------------------------------------------

typedef __bf16 v16bf __attribute__((ext_vector_type(16)));
typedef float  v8f   __attribute__((ext_vector_type(8)));

#define HD   128      // hidden width H
#define SEQL 48       // sequence length
#define FIN  16       // input features
#define NHEADS 4
#define BNEPS 1e-5f

__device__ __forceinline__ v8f wmma_bf16(v16bf a, v16bf b, v8f c) {
  // D = A(16x32 bf16) * B(32x16 bf16) + C(16x16 f32)
  return __builtin_amdgcn_wmma_f32_16x16x32_bf16(false, a, false, b, (short)0, c, false, false);
}

__device__ __forceinline__ int frag_lane(int r, int kin) { return r + 16 * ((kin >> 3) & 1); }
__device__ __forceinline__ int frag_elem(int kin)        { return (kin & 7) + 8 * (kin >> 4); }

// monotone float <-> uint encoding for atomicMax on floats
__device__ __forceinline__ unsigned fenc(float f) {
  unsigned u = __float_as_uint(f);
  return (u & 0x80000000u) ? ~u : (u | 0x80000000u);
}
__device__ __forceinline__ float fdec(unsigned e) {
  unsigned u = (e & 0x80000000u) ? (e ^ 0x80000000u) : ~e;
  return __uint_as_float(u);
}

__device__ __forceinline__ void edge_sd(const int* __restrict__ ei, int E, int idx,
                                        int& s, int& d) {
  if (idx < E) { s = ei[idx]; d = ei[E + idx]; }
  else         { s = d = idx - E; }   // appended self-loops
}

// ---------------------------------------------------------------------------
// Weight prep: fold BN into conv weights, convert to bf16, write A-fragment
// order (M = out-channel tiles).  w is (128, Cin, 3) row-major, K = Cin*3.
// ---------------------------------------------------------------------------
__global__ void prep_convw(const float* __restrict__ w, const float* __restrict__ b,
                           const float* __restrict__ g, const float* __restrict__ be,
                           const float* __restrict__ m, const float* __restrict__ v,
                           __bf16* __restrict__ wf, float* __restrict__ bf,
                           int K, int KB) {
  int idx = blockIdx.x * blockDim.x + threadIdx.x;
  int tot = HD * KB * 32;
  if (idx >= tot) return;
  int co = idx / (KB * 32);
  int kp = idx % (KB * 32);
  float s = g[co] * rsqrtf(v[co] + BNEPS);
  float val = (kp < K) ? w[(size_t)co * K + kp] * s : 0.f;
  int mt = co >> 4, r = co & 15;
  int kb = kp >> 5, kin = kp & 31;
  wf[(size_t)((mt * KB + kb) * 32 + frag_lane(r, kin)) * 16 + frag_elem(kin)] = (__bf16)val;
  if (kp == 0) bf[co] = (b[co] - m[co]) * s + be[co];
}

// Pack a (K, Nout) row-major f32 matrix into B-fragment order (N-tile major).
__global__ void prep_bfrag(const float* __restrict__ W, __bf16* __restrict__ out,
                           int K, int Nout) {
  int idx = blockIdx.x * blockDim.x + threadIdx.x;
  if (idx >= K * Nout) return;
  int k = idx / Nout, nc = idx % Nout;
  int KB = K >> 5;
  int nt = nc >> 4, nl = nc & 15;
  int kb = k >> 5, kin = k & 31;
  out[(size_t)((nt * KB + kb) * 32 + frag_lane(nl, kin)) * 16 + frag_elem(kin)] = (__bf16)W[idx];
}

// ---------------------------------------------------------------------------
// Fused TCN: one workgroup (8 waves) per node.  The node's whole activation
// (128x48 bf16 = 12KB) lives in LDS; per layer: build im2col in B-fragment
// order -> WMMA -> bias+ReLU back to canonical (c,t).  Ends with temporal
// mean + fc1 + fc2 and writes nf in A-fragment order for the GAT1 GEMM.
// ---------------------------------------------------------------------------
__global__ void __launch_bounds__(256) tcn_kernel(
    const float* __restrict__ hs, const float* __restrict__ sat,
    const float* __restrict__ stf,
    const __bf16* __restrict__ wInitF, const float* __restrict__ bInit,
    const __bf16* __restrict__ wDilF,  const float* __restrict__ bDil,
    const float* __restrict__ fc1w, const float* __restrict__ fc1b,
    const float* __restrict__ fc2w, const float* __restrict__ fc2b,
    __bf16* __restrict__ nfF) {
  __shared__ __attribute__((aligned(32))) __bf16 xcan[HD * SEQL];    // canonical (c,t)
  __shared__ __attribute__((aligned(32))) __bf16 im2[12 * 3 * 512];  // B-fragment im2col
  __shared__ float comb[136];
  __shared__ float fcb[HD];

  const int node = blockIdx.x;
  const int tid  = threadIdx.x;
  const int lane = tid & 31;
  const int w    = tid >> 5;
  const int grp  = lane >> 4;
  const int nl   = lane & 15;

  // ---- init conv: K = 16*3 = 48 (padded to KB=2 blocks), same-pad 1 ----
  for (int e = tid; e < 2 * 3 * 512; e += 256) {
    int i = e & 15, ln = (e >> 4) & 31, nt = (e >> 9) % 3, kb = e / 1536;
    int kin = 8 * (ln >> 4) + (i & 7) + 16 * (i >> 3);
    int k = kb * 32 + kin;
    float val = 0.f;
    if (k < 48) {
      int ci = k / 3, dt = k % 3;
      int tt = nt * 16 + (ln & 15) + dt - 1;
      if (tt >= 0 && tt < SEQL) val = hs[(size_t)node * (SEQL * FIN) + tt * FIN + ci];
    }
    im2[e] = (__bf16)val;
  }
  __syncthreads();
  {
    v8f a0 = {}, a1 = {}, a2 = {};
    const v16bf* A = (const v16bf*)wInitF;
#pragma unroll
    for (int kb = 0; kb < 2; ++kb) {
      v16bf af = A[(w * 2 + kb) * 32 + lane];
      v16bf b0 = *(const v16bf*)&im2[(kb * 3 + 0) * 512 + lane * 16];
      v16bf b1 = *(const v16bf*)&im2[(kb * 3 + 1) * 512 + lane * 16];
      v16bf b2 = *(const v16bf*)&im2[(kb * 3 + 2) * 512 + lane * 16];
      a0 = wmma_bf16(af, b0, a0);
      a1 = wmma_bf16(af, b1, a1);
      a2 = wmma_bf16(af, b2, a2);
    }
#pragma unroll
    for (int vv = 0; vv < 8; ++vv) {   // C layout: M = vv + 8*grp, N = nl
      int co = 16 * w + vv + 8 * grp;
      float bi = bInit[co];
      xcan[co * SEQL +  0 + nl] = (__bf16)fmaxf(a0[vv] + bi, 0.f);
      xcan[co * SEQL + 16 + nl] = (__bf16)fmaxf(a1[vv] + bi, 0.f);
      xcan[co * SEQL + 32 + nl] = (__bf16)fmaxf(a2[vv] + bi, 0.f);
    }
  }
  __syncthreads();

  // ---- dilated causal convs: d = 1,2,4, effective left pad 2d ----
  for (int ld = 0; ld < 3; ++ld) {
    const int d = 1 << ld;
    for (int e = tid; e < 12 * 3 * 512; e += 256) {
      int i = e & 15, ln = (e >> 4) & 31, nt = (e >> 9) % 3, kb = e / 1536;
      int kin = 8 * (ln >> 4) + (i & 7) + 16 * (i >> 3);
      int k = kb * 32 + kin;
      int ci = k / 3, dt = k % 3;
      int tt = nt * 16 + (ln & 15) + dt * d - 2 * d;
      im2[e] = (tt >= 0 && tt < SEQL) ? xcan[ci * SEQL + tt] : (__bf16)0.f;
    }
    __syncthreads();
    v8f a0 = {}, a1 = {}, a2 = {};
    const v16bf* A = (const v16bf*)(wDilF + (size_t)ld * 49152);
    for (int kb = 0; kb < 12; ++kb) {
      v16bf af = A[(w * 12 + kb) * 32 + lane];
      v16bf b0 = *(const v16bf*)&im2[(kb * 3 + 0) * 512 + lane * 16];
      v16bf b1 = *(const v16bf*)&im2[(kb * 3 + 1) * 512 + lane * 16];
      v16bf b2 = *(const v16bf*)&im2[(kb * 3 + 2) * 512 + lane * 16];
      a0 = wmma_bf16(af, b0, a0);
      a1 = wmma_bf16(af, b1, a1);
      a2 = wmma_bf16(af, b2, a2);
    }
    const float* bl = bDil + ld * HD;
#pragma unroll
    for (int vv = 0; vv < 8; ++vv) {
      int co = 16 * w + vv + 8 * grp;
      float bi = bl[co];
      xcan[co * SEQL +  0 + nl] = (__bf16)fmaxf(a0[vv] + bi, 0.f);
      xcan[co * SEQL + 16 + nl] = (__bf16)fmaxf(a1[vv] + bi, 0.f);
      xcan[co * SEQL + 32 + nl] = (__bf16)fmaxf(a2[vv] + bi, 0.f);
    }
    __syncthreads();
  }

  // ---- temporal mean + concat(sat, static) ----
  if (tid < HD) {
    float s = 0.f;
    for (int t = 0; t < SEQL; ++t) s += (float)xcan[tid * SEQL + t];
    comb[tid] = s * (1.f / SEQL);
  } else if (tid < HD + 3) {
    comb[tid] = sat[(size_t)node * 3 + (tid - HD)];
  } else if (tid < HD + 5) {
    comb[tid] = stf[(size_t)node * 2 + (tid - HD - 3)];
  }
  __syncthreads();
  if (tid < HD) {
    float a = fc1b[tid];
    for (int i2 = 0; i2 < HD + 5; ++i2) a += comb[i2] * fc1w[i2 * HD + tid];
    fcb[tid] = fmaxf(a, 0.f);
  }
  __syncthreads();
  if (tid < HD) {
    float a = fc2b[tid];
    for (int i2 = 0; i2 < HD; ++i2) a += fcb[i2] * fc2w[i2 * HD + tid];
    a = fmaxf(a, 0.f);
    // write nf in A-fragment order (node tiles of 16, K = 128 -> KB = 4)
    int mt = node >> 4, r = node & 15;
    int kb = tid >> 5, kin = tid & 31;
    nfF[(size_t)((mt * 4 + kb) * 32 + frag_lane(r, kin)) * 16 + frag_elem(kin)] = (__bf16)a;
  }
}

// ---------------------------------------------------------------------------
// Generic fragment GEMM: C(MT*16, Nout) = Afrag(node tiles) x Bfrag(weights).
// One block per M-tile; wave w covers N-tiles w, w+8, ...
// ---------------------------------------------------------------------------
__global__ void __launch_bounds__(256) gemm_frag(
    const __bf16* __restrict__ Afrag, const __bf16* __restrict__ Bfrag,
    float* __restrict__ C, int KB, int NT, int Nout) {
  int mt = blockIdx.x;
  int lane = threadIdx.x & 31, w = threadIdx.x >> 5;
  int grp = lane >> 4, nl = lane & 15;
  const v16bf* A  = (const v16bf*)Afrag + (size_t)mt * KB * 32;
  const v16bf* Bb = (const v16bf*)Bfrag;
  for (int nt = w; nt < NT; nt += 8) {
    const v16bf* B = Bb + (size_t)nt * KB * 32;
    v8f acc = {};
    for (int kb = 0; kb < KB; ++kb) {
      if (kb + 1 < KB) __builtin_prefetch(&B[(kb + 1) * 32 + lane], 0, 0);
      acc = wmma_bf16(A[kb * 32 + lane], B[kb * 32 + lane], acc);
    }
    int row = mt * 16 + 8 * grp;
#pragma unroll
    for (int vv = 0; vv < 8; ++vv)
      C[(size_t)(row + vv) * Nout + nt * 16 + nl] = acc[vv];
  }
}

// per-node attention logits: as[n,h] = <h[n,h,:], a_src[h,:]>, ad likewise
__global__ void attn_dots(const float* __restrict__ h, const float* __restrict__ asrc,
                          const float* __restrict__ adst, float* __restrict__ oas,
                          float* __restrict__ oad, int N, int heads) {
  int idx = blockIdx.x * blockDim.x + threadIdx.x;
  if (idx >= N * heads) return;
  int n = idx / heads, hd = idx % heads;
  const float* hp = h + ((size_t)n * heads + hd) * HD;
  float s1 = 0.f, s2 = 0.f;
  for (int d = 0; d < HD; ++d) {
    s1 += hp[d] * asrc[hd * HD + d];
    s2 += hp[d] * adst[hd * HD + d];
  }
  oas[idx] = s1;
  oad[idx] = s2;
}

__global__ void fill_u32(unsigned* __restrict__ p, unsigned v, int n) {
  int i = blockIdx.x * blockDim.x + threadIdx.x;
  if (i < n) p[i] = v;
}

__global__ void edge_max(const int* __restrict__ ei, int E, int EE,
                         const float* __restrict__ as, const float* __restrict__ ad,
                         unsigned* __restrict__ menc, int heads) {
  int idx = blockIdx.x * blockDim.x + threadIdx.x;
  if (idx >= EE) return;
  int s, d; edge_sd(ei, E, idx, s, d);
  for (int h = 0; h < heads; ++h) {
    float e = as[s * heads + h] + ad[d * heads + h];
    e = e > 0.f ? e : 0.2f * e;                 // leaky_relu(0.2)
    atomicMax(&menc[d * heads + h], fenc(e));
  }
}

__global__ void edge_expsum(const int* __restrict__ ei, int E, int EE,
                            const float* __restrict__ as, const float* __restrict__ ad,
                            const unsigned* __restrict__ menc,
                            float* __restrict__ exb, float* __restrict__ ssum, int heads) {
  int idx = blockIdx.x * blockDim.x + threadIdx.x;
  if (idx >= EE) return;
  int s, d; edge_sd(ei, E, idx, s, d);
  for (int h = 0; h < heads; ++h) {
    float e = as[s * heads + h] + ad[d * heads + h];
    e = e > 0.f ? e : 0.2f * e;
    float ex = __expf(e - fdec(menc[d * heads + h]));
    exb[(size_t)idx * heads + h] = ex;
    atomicAdd(&ssum[d * heads + h], ex);
  }
}

// GAT1 aggregation: 512 dims (4 heads x 128), 64 threads/edge, 8 dims each
__global__ void edge_aggr512(const int* __restrict__ ei, int E, int EE,
                             const float* __restrict__ h, const float* __restrict__ exb,
                             const float* __restrict__ ssum, float* __restrict__ out) {
  int eidx = blockIdx.x * 4 + (threadIdx.x >> 6);
  if (eidx >= EE) return;
  int s, d; edge_sd(ei, E, eidx, s, d);
  int base = (threadIdx.x & 63) * 8;
  int hd = base >> 7;
  float alpha = exb[(size_t)eidx * 4 + hd] / (ssum[d * 4 + hd] + 1e-16f);
#pragma unroll
  for (int j = 0; j < 8; ++j) {
    int dim = base + j;
    atomicAdd(&out[(size_t)d * 512 + dim], h[(size_t)s * 512 + dim] * alpha);
  }
}

// GAT2 aggregation: 128 dims, 1 head, 128 threads/edge
__global__ void edge_aggr128(const int* __restrict__ ei, int E, int EE,
                             const float* __restrict__ h, const float* __restrict__ exb,
                             const float* __restrict__ ssum, float* __restrict__ out) {
  int eidx = blockIdx.x * 2 + (threadIdx.x >> 7);
  if (eidx >= EE) return;
  int s, d; edge_sd(ei, E, eidx, s, d);
  int dim = threadIdx.x & 127;
  float alpha = exb[eidx] / (ssum[d] + 1e-16f);
  atomicAdd(&out[(size_t)d * HD + dim], h[(size_t)s * HD + dim] * alpha);
}

// bias + ELU, pack g into A-fragment order for GAT2 GEMM (K = 512, KB = 16)
__global__ void gat1_post(const float* __restrict__ out1, const float* __restrict__ b,
                          __bf16* __restrict__ gF, int N) {
  int idx = blockIdx.x * blockDim.x + threadIdx.x;
  if (idx >= N * 512) return;
  int n = idx >> 9, dim = idx & 511;
  float g = out1[idx] + b[dim];
  g = g > 0.f ? g : expm1f(g);   // elu
  int mt = n >> 4, r = n & 15;
  int kb = dim >> 5, kin = dim & 31;
  gF[(size_t)((mt * 16 + kb) * 32 + frag_lane(r, kin)) * 16 + frag_elem(kin)] = (__bf16)g;
}

// bias + 3-layer MLP head, one node per 128-thread block
__global__ void __launch_bounds__(128) gat2_post_mlp(
    const float* __restrict__ out2, const float* __restrict__ b2,
    const float* __restrict__ m1w, const float* __restrict__ m1b,
    const float* __restrict__ m2w, const float* __restrict__ m2b,
    const float* __restrict__ m3w, const float* __restrict__ m3b,
    float* __restrict__ out) {
  __shared__ float g2[HD], ha[HD], hb[64];
  int n = blockIdx.x, t = threadIdx.x;
  g2[t] = out2[(size_t)n * HD + t] + b2[t];
  __syncthreads();
  {
    float a = m1b[t];
    for (int i = 0; i < HD; ++i) a += g2[i] * m1w[i * HD + t];
    ha[t] = fmaxf(a, 0.f);
  }
  __syncthreads();
  if (t < 64) {
    float a = m2b[t];
    for (int i = 0; i < HD; ++i) a += ha[i] * m2w[i * 64 + t];
    hb[t] = fmaxf(a, 0.f);
  }
  __syncthreads();
  if (t < 2) {
    float a = m3b[t];
    for (int i = 0; i < 64; ++i) a += hb[i] * m3w[i * 2 + t];
    out[(size_t)n * 2 + t] = a;
  }
}

// ---------------------------------------------------------------------------
extern "C" void kernel_launch(void* const* d_in, const int* in_sizes, int n_in,
                              void* d_out, int out_size, void* d_ws, size_t ws_size,
                              hipStream_t stream) {
  // dict order: hourly_seq, daily_satellite, static_features, edge_index,
  // then params leaves in pytree (sorted-key) order.
  const float* hs  = (const float*)d_in[0];
  const float* sat = (const float*)d_in[1];
  const float* stf = (const float*)d_in[2];
  const int*   ei  = (const int*)d_in[3];
  const float* bn0_g  = (const float*)d_in[4];
  const float* bn0_be = (const float*)d_in[5];
  const float* bn0_m  = (const float*)d_in[6];
  const float* bn0_v  = (const float*)d_in[7];
  const float *dil_b[3], *dil_g[3], *dil_be[3], *dil_m[3], *dil_v[3], *dil_w[3];
  for (int l = 0; l < 3; ++l) {
    dil_b[l]  = (const float*)d_in[8 + 6 * l];
    dil_g[l]  = (const float*)d_in[9 + 6 * l];
    dil_be[l] = (const float*)d_in[10 + 6 * l];
    dil_m[l]  = (const float*)d_in[11 + 6 * l];
    dil_v[l]  = (const float*)d_in[12 + 6 * l];
    dil_w[l]  = (const float*)d_in[13 + 6 * l];
  }
  const float* fc1_b = (const float*)d_in[26];
  const float* fc1_w = (const float*)d_in[27];
  const float* fc2_b = (const float*)d_in[28];
  const float* fc2_w = (const float*)d_in[29];
  const float* g1_adst = (const float*)d_in[30];
  const float* g1_asrc = (const float*)d_in[31];
  const float* g1_b    = (const float*)d_in[32];
  const float* g1_w    = (const float*)d_in[33];
  const float* g2_adst = (const float*)d_in[34];
  const float* g2_asrc = (const float*)d_in[35];
  const float* g2_b    = (const float*)d_in[36];
  const float* g2_w    = (const float*)d_in[37];
  const float* init_b  = (const float*)d_in[38];
  const float* init_w  = (const float*)d_in[39];
  const float* m1_b = (const float*)d_in[40];
  const float* m1_w = (const float*)d_in[41];
  const float* m2_b = (const float*)d_in[42];
  const float* m2_w = (const float*)d_in[43];
  const float* m3_b = (const float*)d_in[44];
  const float* m3_w = (const float*)d_in[45];

  const int N  = in_sizes[0] / (SEQL * FIN);
  const int E  = in_sizes[3] / 2;
  const int EE = E + N;        // + self-loops
  const int MT = N / 16;
  (void)n_in; (void)out_size; (void)ws_size;

  // workspace bump allocator
  char* p = (char*)d_ws;
  auto alloc = [&](size_t bytes) -> char* {
    char* r = p;
    p += (bytes + 255) & ~(size_t)255;
    return r;
  };
  __bf16* wInitF = (__bf16*)alloc((size_t)8 * 2 * 512 * 2);
  __bf16* wDilF  = (__bf16*)alloc((size_t)3 * 8 * 12 * 512 * 2);
  float*  bInit  = (float*)alloc(128 * 4);
  float*  bDil   = (float*)alloc(3 * 128 * 4);
  __bf16* w1F    = (__bf16*)alloc((size_t)32 * 4 * 512 * 2);
  __bf16* w2F    = (__bf16*)alloc((size_t)8 * 16 * 512 * 2);
  __bf16* nfF    = (__bf16*)alloc((size_t)MT * 4 * 512 * 2);
  float*  h1     = (float*)alloc((size_t)N * 512 * 4);
  float*  as1    = (float*)alloc((size_t)N * NHEADS * 4);
  float*  ad1    = (float*)alloc((size_t)N * NHEADS * 4);
  unsigned* mx1  = (unsigned*)alloc((size_t)N * NHEADS * 4);
  float*  s1     = (float*)alloc((size_t)N * NHEADS * 4);
  float*  ex1    = (float*)alloc((size_t)EE * NHEADS * 4);
  float*  out1   = (float*)alloc((size_t)N * 512 * 4);
  __bf16* gF     = (__bf16*)alloc((size_t)MT * 16 * 512 * 2);
  float*  h2     = (float*)alloc((size_t)N * HD * 4);
  float*  as2    = (float*)alloc((size_t)N * 4);
  float*  ad2    = (float*)alloc((size_t)N * 4);
  unsigned* mx2  = (unsigned*)alloc((size_t)N * 4);
  float*  s2     = (float*)alloc((size_t)N * 4);
  float*  ex2    = (float*)alloc((size_t)EE * 4);
  float*  out2   = (float*)alloc((size_t)N * HD * 4);

  const unsigned NEG_INF_ENC = 0x007FFFFFu;   // fenc(-inf)

  // ---- weight prep ----
  prep_convw<<<(HD * 2 * 32 + 255) / 256, 256, 0, stream>>>(
      init_w, init_b, bn0_g, bn0_be, bn0_m, bn0_v, wInitF, bInit, 48, 2);
  for (int l = 0; l < 3; ++l)
    prep_convw<<<(HD * 12 * 32 + 255) / 256, 256, 0, stream>>>(
        dil_w[l], dil_b[l], dil_g[l], dil_be[l], dil_m[l], dil_v[l],
        wDilF + (size_t)l * 49152, bDil + l * 128, 384, 12);
  prep_bfrag<<<(128 * 512 + 255) / 256, 256, 0, stream>>>(g1_w, w1F, 128, 512);
  prep_bfrag<<<(512 * 128 + 255) / 256, 256, 0, stream>>>(g2_w, w2F, 512, 128);

  // ---- fused TCN + fc1/fc2 (one WG per node) ----
  tcn_kernel<<<N, 256, 0, stream>>>(hs, sat, stf, wInitF, bInit, wDilF, bDil,
                                    fc1_w, fc1_b, fc2_w, fc2_b, nfF);

  // ---- GAT layer 1 ----
  gemm_frag<<<MT, 256, 0, stream>>>(nfF, w1F, h1, 4, 32, 512);
  attn_dots<<<(N * NHEADS + 255) / 256, 256, 0, stream>>>(h1, g1_asrc, g1_adst,
                                                          as1, ad1, N, NHEADS);
  fill_u32<<<(N * NHEADS + 255) / 256, 256, 0, stream>>>(mx1, NEG_INF_ENC, N * NHEADS);
  (void)hipMemsetAsync(s1, 0, (size_t)N * NHEADS * 4, stream);
  (void)hipMemsetAsync(out1, 0, (size_t)N * 512 * 4, stream);
  edge_max<<<(EE + 255) / 256, 256, 0, stream>>>(ei, E, EE, as1, ad1, mx1, NHEADS);
  edge_expsum<<<(EE + 255) / 256, 256, 0, stream>>>(ei, E, EE, as1, ad1, mx1,
                                                    ex1, s1, NHEADS);
  edge_aggr512<<<(EE + 3) / 4, 256, 0, stream>>>(ei, E, EE, h1, ex1, s1, out1);
  gat1_post<<<(N * 512 + 255) / 256, 256, 0, stream>>>(out1, g1_b, gF, N);

  // ---- GAT layer 2 ----
  gemm_frag<<<MT, 256, 0, stream>>>(gF, w2F, h2, 16, 8, 128);
  attn_dots<<<(N + 255) / 256, 256, 0, stream>>>(h2, g2_asrc, g2_adst, as2, ad2, N, 1);
  fill_u32<<<(N + 255) / 256, 256, 0, stream>>>(mx2, NEG_INF_ENC, N);
  (void)hipMemsetAsync(s2, 0, (size_t)N * 4, stream);
  (void)hipMemsetAsync(out2, 0, (size_t)N * HD * 4, stream);
  edge_max<<<(EE + 255) / 256, 256, 0, stream>>>(ei, E, EE, as2, ad2, mx2, 1);
  edge_expsum<<<(EE + 255) / 256, 256, 0, stream>>>(ei, E, EE, as2, ad2, mx2,
                                                    ex2, s2, 1);
  edge_aggr128<<<(EE + 1) / 2, 256, 0, stream>>>(ei, E, EE, h2, ex2, s2, out2);

  // ---- MLP head ----
  gat2_post_mlp<<<N, 128, 0, stream>>>(out2, g2_b, m1_w, m1_b, m2_w, m2_b,
                                       m3_w, m3_b, (float*)d_out);
}
